// MultiHeadLatentAttention_61237643706963
// MI455X (gfx1250) — matrix operand order repaired and checked by reading, not compiled, not run on hardware
//
#include <hip/hip_runtime.h>
#include <hip/hip_bf16.h>

// ---------------- problem constants ----------------
constexpr int Bv  = 2;
constexpr int Tv  = 2048;
constexpr int DIMv= 2048;
constexpr int Hv  = 16;
constexpr int DHv = 128;
constexpr int QCv = 512;
constexpr int KVCv= 512;
constexpr int DDv = 64;            // rope dims
constexpr int HDK = DHv + DDv;     // 192 per-head q/k width
constexpr int MROWS = Bv * Tv;     // 4096 "token" rows

typedef __attribute__((ext_vector_type(16))) __bf16 bf16x16;
typedef __attribute__((ext_vector_type(8)))  float  f32x8;

struct u4 { unsigned x, y, z, w; };   // POD 16-byte chunk

__device__ __forceinline__ unsigned short f2bf(float f) {
    union { float f; unsigned u; } v; v.f = f;
    unsigned r = v.u + 0x7FFFu + ((v.u >> 16) & 1u);   // RNE
    return (unsigned short)(r >> 16);
}
__device__ __forceinline__ float bf2f(unsigned short s) {
    union { unsigned u; float f; } v; v.u = ((unsigned)s) << 16;
    return v.f;
}
__device__ __forceinline__ f32x8 zero8() {
    f32x8 z; for (int i = 0; i < 8; ++i) z[i] = 0.0f; return z;
}

// CDNA5 async global->LDS copy (16B per lane), tracked by ASYNCcnt.
// LDS destination = low 32 bits of the generic pointer.
__device__ __forceinline__ void async_g2l_b128(void* lds_ptr, const void* g_ptr) {
#if defined(__HIP_DEVICE_COMPILE__)
    unsigned lds_off = (unsigned)(unsigned long long)lds_ptr;
    unsigned long long ga = (unsigned long long)g_ptr;
    asm volatile("global_load_async_to_lds_b128 %0, %1, off"
                 :: "v"(lds_off), "v"(ga) : "memory");
#else
    (void)lds_ptr; (void)g_ptr;
#endif
}
__device__ __forceinline__ void wait_asynccnt0() {
#if defined(__HIP_DEVICE_COMPILE__)
#if __has_builtin(__builtin_amdgcn_s_wait_asynccnt)
    __builtin_amdgcn_s_wait_asynccnt(0);
#else
    asm volatile("s_wait_asynccnt 0" ::: "memory");
#endif
#endif
}

// ---------------- fp32 -> bf16 convert ----------------
__global__ void cvt_bf16_kernel(const float* __restrict__ in,
                                unsigned short* __restrict__ out, long long n) {
    long long i = (long long)blockIdx.x * blockDim.x + threadIdx.x;
    long long stride = (long long)gridDim.x * blockDim.x;
    for (; i < n; i += stride) out[i] = f2bf(in[i]);
}

// ---------------- tiled bf16 WMMA GEMM ----------------
// C[M,N] = A[M,K] @ B[K,N]; row-major; M%64==0, N%64==0, K%32==0.
// 128 threads = 4 waves in a 2x2 grid; each wave computes 32x32 via 2x2 WMMA tiles.
template <int OUTF32>
__global__ __launch_bounds__(128)
void gemm_bf16_kernel(const unsigned short* __restrict__ A,
                      const unsigned short* __restrict__ Bm,
                      void* __restrict__ Cv,
                      int M, int N, int K) {
    __shared__ __align__(16) unsigned short lA[64][32];   // A tile, row-major
    __shared__ __align__(16) unsigned short lBt[64][32];  // B tile, transposed [n][k]

    const int tid  = threadIdx.x;
    const int lane = tid & 31;
    const int wid  = tid >> 5;
    const int mtile = blockIdx.y * 64;
    const int ntile = blockIdx.x * 64;
    const int wm = (wid & 1) * 32;
    const int wn = (wid >> 1) * 32;

    const int nlo = lane & 15;
    const int kbA = (lane & 16) ? 8 : 0;    // A-frag K sub-offset
    const int kbB = (lane & 16) ? 8 : 0;    // B-frag K sub-offset

    f32x8 acc[2][2];
    for (int i = 0; i < 2; ++i) for (int j = 0; j < 2; ++j) acc[i][j] = zero8();

    for (int k0 = 0; k0 < K; k0 += 32) {
        __syncthreads();   // protect previous iteration's LDS reads
        // stage A: 64x32 bf16 -> 256 16B chunks (async global->LDS)
        for (int c = tid; c < 256; c += 128) {
            int row = c >> 2, col8 = (c & 3) * 8;
            async_g2l_b128(&lA[row][col8],
                           &A[(size_t)(mtile + row) * K + k0 + col8]);
        }
        // stage B transposed: 32x64 bf16, contiguous 8-half reads, scatter to [n][k]
        for (int c = tid; c < 256; c += 128) {
            int kr = c >> 3, col8 = (c & 7) * 8;
            u4 t = *(const u4*)&Bm[(size_t)(k0 + kr) * N + ntile + col8];
            const unsigned short* h = (const unsigned short*)&t;
            for (int e = 0; e < 8; ++e) lBt[col8 + e][kr] = h[e];
        }
        wait_asynccnt0();
        __syncthreads();

        bf16x16 af[2], bfg[2];
        for (int s = 0; s < 2; ++s) {
            int ar = wm + s * 16 + nlo;
            u4* p = (u4*)&af[s];
            p[0] = *(const u4*)&lA[ar][kbA];
            p[1] = *(const u4*)&lA[ar][16 + kbA];
        }
        for (int s = 0; s < 2; ++s) {
            int bn = wn + s * 16 + nlo;
            u4* p = (u4*)&bfg[s];
            p[0] = *(const u4*)&lBt[bn][kbB];
            p[1] = *(const u4*)&lBt[bn][kbB + 16];
        }
        for (int i = 0; i < 2; ++i)
            for (int j = 0; j < 2; ++j)
                acc[i][j] = __builtin_amdgcn_wmma_f32_16x16x32_bf16(
                    false, af[i], false, bfg[j], (short)0, acc[i][j], false, false);
    }

    const int rsel = (lane & 16) ? 8 : 0;
    for (int i = 0; i < 2; ++i)
        for (int j = 0; j < 2; ++j)
            for (int r = 0; r < 8; ++r) {
                int gm = mtile + wm + i * 16 + r + rsel;
                int gn = ntile + wn + j * 16 + nlo;
                float v = acc[i][j][r];
                if (OUTF32) ((float*)Cv)[(size_t)gm * N + gn] = v;
                else        ((unsigned short*)Cv)[(size_t)gm * N + gn] = f2bf(v);
            }
}

// ---------------- q/k head assembly + RoPE ----------------
__global__ __launch_bounds__(128)
void build_q_kernel(const unsigned short* __restrict__ qu,     // [M, H*128]
                    const unsigned short* __restrict__ qr,     // [M, H*64]
                    const float* __restrict__ freqs,           // [T, 32, 2]
                    unsigned short* __restrict__ qh) {         // [B,H,T,192]
    int idx = blockIdx.x;
    int h = idx % Hv; int t = (idx / Hv) % Tv; int b = idx / (Hv * Tv);
    size_t row = (size_t)b * Tv + t;
    size_t obase = (((size_t)b * Hv + h) * Tv + t) * HDK;
    int d = threadIdx.x;
    qh[obase + d] = qu[row * (Hv * DHv) + h * DHv + d];
    if (d < 32) {
        float x1 = bf2f(qr[row * (Hv * DDv) + h * DDv + 2 * d]);
        float x2 = bf2f(qr[row * (Hv * DDv) + h * DDv + 2 * d + 1]);
        float c = freqs[((size_t)t * 32 + d) * 2 + 0];
        float s = freqs[((size_t)t * 32 + d) * 2 + 1];
        qh[obase + DHv + 2 * d]     = f2bf(x1 * c - x2 * s);
        qh[obase + DHv + 2 * d + 1] = f2bf(x1 * s + x2 * c);
    }
}

__global__ __launch_bounds__(128)
void build_k_kernel(const unsigned short* __restrict__ ku,     // [M, H*128]
                    const unsigned short* __restrict__ kr,     // [M, 64] shared across heads
                    const float* __restrict__ freqs,
                    unsigned short* __restrict__ kh) {         // [B,H,T,192]
    int idx = blockIdx.x;
    int h = idx % Hv; int t = (idx / Hv) % Tv; int b = idx / (Hv * Tv);
    size_t row = (size_t)b * Tv + t;
    size_t obase = (((size_t)b * Hv + h) * Tv + t) * HDK;
    int d = threadIdx.x;
    kh[obase + d] = ku[row * (Hv * DHv) + h * DHv + d];
    if (d < 32) {
        float x1 = bf2f(kr[row * DDv + 2 * d]);
        float x2 = bf2f(kr[row * DDv + 2 * d + 1]);
        float c = freqs[((size_t)t * 32 + d) * 2 + 0];
        float s = freqs[((size_t)t * 32 + d) * 2 + 1];
        kh[obase + DHv + 2 * d]     = f2bf(x1 * c - x2 * s);
        kh[obase + DHv + 2 * d + 1] = f2bf(x1 * s + x2 * c);
    }
}

__global__ void build_v_kernel(const unsigned short* __restrict__ vu,  // [M, H*128]
                               unsigned short* __restrict__ vh) {      // [B,H,T,128]
    long long n = (long long)Bv * Tv * Hv * DHv;
    long long i = (long long)blockIdx.x * blockDim.x + threadIdx.x;
    long long stride = (long long)gridDim.x * blockDim.x;
    for (; i < n; i += stride) {
        int d = (int)(i % DHv);
        int h = (int)((i / DHv) % Hv);
        int t = (int)((i / ((long long)DHv * Hv)) % Tv);
        int b = (int)(i / ((long long)DHv * Hv * Tv));
        vh[(((size_t)b * Hv + h) * Tv + t) * DHv + d] =
            vu[((size_t)b * Tv + t) * (Hv * DHv) + h * DHv + d];
    }
}

// ---------------- flash attention with WMMA ----------------
// grid.x = T/64 (q tiles), grid.y = B*H. 128 threads = 4 waves; wave w owns 16 q rows.
__global__ __launch_bounds__(128)
void attn_kernel(const unsigned short* __restrict__ qh,   // [B,H,T,192]
                 const unsigned short* __restrict__ kh,   // [B,H,T,192]
                 const unsigned short* __restrict__ vh,   // [B,H,T,128]
                 unsigned short* __restrict__ ao) {       // [M, H*128]
    __shared__ __align__(16) unsigned short Kl[64][HDK];      // K tile (async-staged)
    __shared__ __align__(16) unsigned short Vt[DHv][64];      // V tile transposed [d][key]
    __shared__ __align__(16) unsigned short Pl[4][16 * 64];   // per-wave P scratch

    const int tid  = threadIdx.x;
    const int lane = tid & 31;
    const int wid  = tid >> 5;
    const int bh = blockIdx.y;
    const int b  = bh / Hv;
    const int h  = bh % Hv;
    const int qtile = blockIdx.x * 64;
    const int qrow0 = qtile + wid * 16;

    const unsigned short* Q  = qh + (size_t)bh * Tv * HDK;
    const unsigned short* Kp = kh + (size_t)bh * Tv * HDK;
    const unsigned short* Vp = vh + (size_t)bh * Tv * DHv;

    const int nlo = lane & 15;
    const int kbA = (lane & 16) ? 8 : 0;
    const int kbB = (lane & 16) ? 8 : 0;
    const int rsel = (lane & 16) ? 8 : 0;

    // preload 6 Q A-fragments (192 = 6 * 32)
    bf16x16 qf[6];
    {
        int arow = qrow0 + nlo;
        const unsigned short* qrowp = Q + (size_t)arow * HDK;
        for (int dc = 0; dc < 6; ++dc) {
            u4* p = (u4*)&qf[dc];
            p[0] = *(const u4*)&qrowp[dc * 32 + kbA];
            p[1] = *(const u4*)&qrowp[dc * 32 + 16 + kbA];
        }
    }

    f32x8 o[8];
    for (int dc = 0; dc < 8; ++dc) o[dc] = zero8();
    float mstat[8], lstat[8];
    for (int r = 0; r < 8; ++r) { mstat[r] = -3.0e38f; lstat[r] = 0.0f; }

    const float scale = 0.07216878364870323f;  // 1/sqrt(192)

    for (int kt = 0; kt <= (int)blockIdx.x; ++kt) {
        const int kbase = kt * 64;

        __syncthreads();   // everyone done with previous Kl/Vt/Pl reads
        // --- stage K tile [64][192] via CDNA5 async global->LDS (12 chunks/thread) ---
        for (int j = 0; j < 12; ++j) {
            int c = tid + j * 128;                // 1536 chunks of 8 halves
            int row = c / 24, col8 = (c % 24) * 8;
            async_g2l_b128(&Kl[row][col8],
                           &Kp[(size_t)(kbase + row) * HDK + col8]);
        }
        // --- stage V transposed [d][key] (8 chunks/thread, manual scatter) ---
        for (int j = 0; j < 8; ++j) {
            int c = tid + j * 128;                // 1024 chunks of 8 halves
            int key = c >> 4, col8 = (c & 15) * 8;
            u4 t = *(const u4*)&Vp[(size_t)(kbase + key) * DHv + col8];
            const unsigned short* hh = (const unsigned short*)&t;
            for (int e = 0; e < 8; ++e) Vt[col8 + e][key] = hh[e];
        }
        wait_asynccnt0();
        __syncthreads();

        // --- scores S = Q K^T : 16x64 per wave (batched frag loads, then WMMA burst) ---
        f32x8 sc[4];
        for (int nc = 0; nc < 4; ++nc) sc[nc] = zero8();
        for (int nc = 0; nc < 4; ++nc) {
            int krl = nc * 16 + nlo;              // key row within tile
            bf16x16 kf[6];
            for (int dc = 0; dc < 6; ++dc) {
                u4* p = (u4*)&kf[dc];
                p[0] = *(const u4*)&Kl[krl][dc * 32 + kbB];
                p[1] = *(const u4*)&Kl[krl][dc * 32 + kbB + 16];
            }
            for (int dc = 0; dc < 6; ++dc)
                sc[nc] = __builtin_amdgcn_wmma_f32_16x16x32_bf16(
                    false, qf[dc], false, kf[dc], (short)0, sc[nc], false, false);
        }

        // --- scale + causal mask + online softmax ---
        for (int r = 0; r < 8; ++r) {
            int qglob = qrow0 + r + rsel;
            float rm = -3.0e38f;
            for (int nc = 0; nc < 4; ++nc) {
                int key = kbase + nc * 16 + nlo;
                float s = sc[nc][r] * scale + (key <= qglob ? 0.0f : -1.0e9f);
                sc[nc][r] = s;
                rm = fmaxf(rm, s);
            }
            rm = fmaxf(rm, __shfl_xor(rm, 1, 32));
            rm = fmaxf(rm, __shfl_xor(rm, 2, 32));
            rm = fmaxf(rm, __shfl_xor(rm, 4, 32));
            rm = fmaxf(rm, __shfl_xor(rm, 8, 32));
            float mn = fmaxf(mstat[r], rm);
            float alpha = __expf(mstat[r] - mn);
            mstat[r] = mn;
            float rs = 0.0f;
            for (int nc = 0; nc < 4; ++nc) {
                float p = __expf(sc[nc][r] - mn);
                sc[nc][r] = p;
                rs += p;
            }
            rs += __shfl_xor(rs, 1, 32);
            rs += __shfl_xor(rs, 2, 32);
            rs += __shfl_xor(rs, 4, 32);
            rs += __shfl_xor(rs, 8, 32);
            lstat[r] = lstat[r] * alpha + rs;
            for (int dc = 0; dc < 8; ++dc) o[dc][r] *= alpha;
        }

        // --- P (C-layout) -> per-wave LDS scratch; wave-local, DS ops are in-order ---
        for (int nc = 0; nc < 4; ++nc)
            for (int r = 0; r < 8; ++r) {
                int rl = r + rsel;
                Pl[wid][rl * 64 + nc * 16 + nlo] = f2bf(sc[nc][r]);
            }

        // --- O += P @ V ---
        bf16x16 pf[2];
        for (int kc = 0; kc < 2; ++kc) {
            const unsigned short* prow = &Pl[wid][nlo * 64];
            u4* p = (u4*)&pf[kc];
            p[0] = *(const u4*)&prow[kc * 32 + kbA];
            p[1] = *(const u4*)&prow[kc * 32 + 16 + kbA];
        }
        for (int dc = 0; dc < 8; ++dc) {
            int d = dc * 16 + nlo;
            bf16x16 vf[2];
            for (int kc = 0; kc < 2; ++kc) {
                u4* p = (u4*)&vf[kc];
                p[0] = *(const u4*)&Vt[d][kc * 32 + kbB];
                p[1] = *(const u4*)&Vt[d][kc * 32 + kbB + 16];
            }
            for (int kc = 0; kc < 2; ++kc)
                o[dc] = __builtin_amdgcn_wmma_f32_16x16x32_bf16(
                    false, pf[kc], false, vf[kc], (short)0, o[dc], false, false);
        }
    }

    // --- normalize + write out [M, H*128] ---
    for (int dc = 0; dc < 8; ++dc)
        for (int r = 0; r < 8; ++r) {
            int qglob = qrow0 + r + rsel;
            int d = dc * 16 + nlo;
            float v = o[dc][r] / lstat[r];
            ao[((size_t)b * Tv + qglob) * (Hv * DHv) + h * DHv + d] = f2bf(v);
        }
}

// ---------------- launch ----------------
extern "C" void kernel_launch(void* const* d_in, const int* in_sizes, int n_in,
                              void* d_out, int out_size, void* d_ws, size_t ws_size,
                              hipStream_t stream) {
    const float* x     = (const float*)d_in[0];
    const float* freqs = (const float*)d_in[1];
    // d_in[2] = dense causal mask (recomputed analytically in-kernel)
    const float* W_dq  = (const float*)d_in[3];
    const float* W_uq  = (const float*)d_in[4];
    const float* W_dkv = (const float*)d_in[5];
    const float* W_uk  = (const float*)d_in[6];
    const float* W_uv  = (const float*)d_in[7];
    const float* W_qr  = (const float*)d_in[8];
    const float* W_kr  = (const float*)d_in[9];
    const float* W_o   = (const float*)d_in[10];

    unsigned short* ws = (unsigned short*)d_ws;
    size_t cur = 0;
    auto take = [&](size_t nelem) { unsigned short* p = ws + cur; cur += nelem; return p; };

    unsigned short* xb   = take((size_t)MROWS * DIMv);
    unsigned short* wdq  = take((size_t)DIMv * QCv);
    unsigned short* wuq  = take((size_t)QCv * Hv * DHv);
    unsigned short* wdkv = take((size_t)DIMv * KVCv);
    unsigned short* wuk  = take((size_t)KVCv * Hv * DHv);
    unsigned short* wuv  = take((size_t)KVCv * Hv * DHv);
    unsigned short* wqr  = take((size_t)QCv * Hv * DDv);
    unsigned short* wkr  = take((size_t)DIMv * DDv);
    unsigned short* wo   = take((size_t)Hv * DHv * DIMv);
    unsigned short* qc   = take((size_t)MROWS * QCv);
    unsigned short* kvc  = take((size_t)MROWS * KVCv);
    unsigned short* qr   = take((size_t)MROWS * Hv * DDv);
    unsigned short* kr   = take((size_t)MROWS * DDv);
    unsigned short* qu   = take((size_t)MROWS * Hv * DHv);
    unsigned short* ku   = take((size_t)MROWS * Hv * DHv);
    unsigned short* vu   = take((size_t)MROWS * Hv * DHv);
    unsigned short* qhb  = take((size_t)Bv * Hv * Tv * HDK);
    unsigned short* khb  = take((size_t)Bv * Hv * Tv * HDK);
    unsigned short* vhb  = take((size_t)Bv * Hv * Tv * DHv);
    unsigned short* aob  = take((size_t)MROWS * Hv * DHv);
    (void)ws_size;

    auto cvt = [&](const float* in, unsigned short* out, long long n) {
        int blocks = (int)((n + 255) / 256);
        if (blocks > 8192) blocks = 8192;
        cvt_bf16_kernel<<<blocks, 256, 0, stream>>>(in, out, n);
    };
    cvt(x,     xb,   (long long)MROWS * DIMv);
    cvt(W_dq,  wdq,  (long long)DIMv * QCv);
    cvt(W_uq,  wuq,  (long long)QCv * Hv * DHv);
    cvt(W_dkv, wdkv, (long long)DIMv * KVCv);
    cvt(W_uk,  wuk,  (long long)KVCv * Hv * DHv);
    cvt(W_uv,  wuv,  (long long)KVCv * Hv * DHv);
    cvt(W_qr,  wqr,  (long long)QCv * Hv * DDv);
    cvt(W_kr,  wkr,  (long long)DIMv * DDv);
    cvt(W_o,   wo,   (long long)Hv * DHv * DIMv);

    auto gemm16 = [&](const unsigned short* A, const unsigned short* Bm,
                      unsigned short* C, int M, int N, int K) {
        gemm_bf16_kernel<0><<<dim3(N / 64, M / 64), 128, 0, stream>>>(A, Bm, (void*)C, M, N, K);
    };

    gemm16(xb,  wdq,  qc,  MROWS, QCv,        DIMv);   // q_c
    gemm16(xb,  wdkv, kvc, MROWS, KVCv,       DIMv);   // kv_c
    gemm16(xb,  wkr,  kr,  MROWS, DDv,        DIMv);   // k_rope base
    gemm16(qc,  wqr,  qr,  MROWS, Hv * DDv,   QCv);    // q_rope
    gemm16(qc,  wuq,  qu,  MROWS, Hv * DHv,   QCv);    // q up
    gemm16(kvc, wuk,  ku,  MROWS, Hv * DHv,   KVCv);   // k up
    gemm16(kvc, wuv,  vu,  MROWS, Hv * DHv,   KVCv);   // v up

    build_q_kernel<<<Bv * Tv * Hv, 128, 0, stream>>>(qu, qr, freqs, qhb);
    build_k_kernel<<<Bv * Tv * Hv, 128, 0, stream>>>(ku, kr, freqs, khb);
    build_v_kernel<<<8192, 256, 0, stream>>>(vu, vhb);

    attn_kernel<<<dim3(Tv / 64, Bv * Hv), 128, 0, stream>>>(qhb, khb, vhb, aob);

    // final projection, fp32 output
    gemm_bf16_kernel<1><<<dim3(DIMv / 64, MROWS / 64), 128, 0, stream>>>(
        aob, wo, d_out, MROWS, DIMv, Hv * DHv);

    (void)in_sizes; (void)n_in; (void)out_size;
}
